// SwinTransformerBlock_53979148976451
// MI455X (gfx1250) — matrix operand order repaired and checked by reading, not compiled, not run on hardware
//
#include <hip/hip_runtime.h>
#include <hip/hip_bf16.h>

// Swin block on gfx1250 via v_wmma_f32_16x16x32_bf16.
// B=16, H=W=112, C=192, NH=12, HD=16, WS=7, SHIFT=3, HID=768
// Rows M = 16*112*112 = 200704 (divisible by 16), windows = 4096, N_tok = 49.

typedef unsigned short u16;
typedef __attribute__((ext_vector_type(16))) __bf16 v16bf;
typedef __attribute__((ext_vector_type(8)))  float  v8f;

union Frag {
    v16bf v;
    u16   s[16];
    uint4 q[2];
};

__device__ __forceinline__ u16 f2bf(float f) {
    union { float f; unsigned u; } x; x.f = f;
    unsigned r = x.u + 0x7FFFu + ((x.u >> 16) & 1u);
    return (u16)(r >> 16);
}

// ---------------------------------------------------------------------------
// Weight convert + transpose: in (K,N) f32 row-major -> out (N,K) bf16
// ---------------------------------------------------------------------------
__global__ void __launch_bounds__(256)
transpose_bf16(const float* __restrict__ in, u16* __restrict__ out, int K, int N)
{
    int idx = blockIdx.x * 256 + threadIdx.x;
    if (idx >= K * N) return;
    int n = idx / K, k = idx % K;
    out[idx] = f2bf(in[(size_t)k * N + n]);
}

// ---------------------------------------------------------------------------
// LN1 + cyclic shift(-3,-3) + window partition -> xw bf16 (4096*49, 192)
// one wave per token
// ---------------------------------------------------------------------------
__global__ void __launch_bounds__(256)
ln1_shift_part(const float* __restrict__ x, const float* __restrict__ g,
               const float* __restrict__ bt, u16* __restrict__ xw)
{
    int wid  = blockIdx.x * 8 + (threadIdx.x >> 5);
    int lane = threadIdx.x & 31;
    if (wid >= 200704) return;
    int b = wid / 12544, l = wid - b * 12544;
    const float* xp = x + (size_t)wid * 192;
    float vals[6], s = 0.f, ss = 0.f;
#pragma unroll
    for (int i = 0; i < 6; i++) {
        float v = xp[lane + i * 32];
        vals[i] = v; s += v; ss += v * v;
    }
#pragma unroll
    for (int o = 16; o > 0; o >>= 1) {
        s  += __shfl_xor(s,  o, 32);
        ss += __shfl_xor(ss, o, 32);
    }
    float mean = s * (1.f / 192.f);
    float var  = ss * (1.f / 192.f) - mean * mean;
    float inv  = rsqrtf(var + 1e-5f);
    int h = l / 112, w = l - h * 112;
    int hs = h - 3; if (hs < 0) hs += 112;
    int ws = w - 3; if (ws < 0) ws += 112;
    int win = b * 256 + (hs / 7) * 16 + (ws / 7);
    int n   = (hs % 7) * 7 + (ws % 7);
    u16* op = xw + ((size_t)win * 49 + n) * 192;
#pragma unroll
    for (int i = 0; i < 6; i++) {
        int c = lane + i * 32;
        op[c] = f2bf((vals[i] - mean) * inv * g[c] + bt[c]);
    }
}

// ---------------------------------------------------------------------------
// LN2: x1 f32 (200704,192) -> bf16, row-major
// ---------------------------------------------------------------------------
__global__ void __launch_bounds__(256)
ln2_kernel(const float* __restrict__ x1, const float* __restrict__ g,
           const float* __restrict__ bt, u16* __restrict__ out)
{
    int wid  = blockIdx.x * 8 + (threadIdx.x >> 5);
    int lane = threadIdx.x & 31;
    if (wid >= 200704) return;
    const float* xp = x1 + (size_t)wid * 192;
    float vals[6], s = 0.f, ss = 0.f;
#pragma unroll
    for (int i = 0; i < 6; i++) {
        float v = xp[lane + i * 32];
        vals[i] = v; s += v; ss += v * v;
    }
#pragma unroll
    for (int o = 16; o > 0; o >>= 1) {
        s  += __shfl_xor(s,  o, 32);
        ss += __shfl_xor(ss, o, 32);
    }
    float mean = s * (1.f / 192.f);
    float var  = ss * (1.f / 192.f) - mean * mean;
    float inv  = rsqrtf(var + 1e-5f);
    u16* op = out + (size_t)wid * 192;
#pragma unroll
    for (int i = 0; i < 6; i++) {
        int c = lane + i * 32;
        op[c] = f2bf((vals[i] - mean) * inv * g[c] + bt[c]);
    }
}

// ---------------------------------------------------------------------------
// WMMA GEMM: D(M,N) = A(M,K)bf16 @ Bt(N,K)bf16 + bias, epilogue by EPI.
// One wave computes a 16x64 strip (A fragment reused 4x).
// EPI 0: store bf16 at row*ldc+col         (QKV, ldc=576 interleave)
// EPI 1: exact GELU, store bf16            (fc1)
// EPI 2: proj: window-reverse + roll(+3) scatter, += residual, f32 out
// EPI 3: += residual row-major, f32 out    (fc2 -> d_out in place)
// ---------------------------------------------------------------------------
template <int EPI>
__global__ void __launch_bounds__(256)
gemm_wmma(const u16* __restrict__ A, const u16* __restrict__ Bt,
          const float* __restrict__ bias, const float* __restrict__ resid,
          u16* __restrict__ outb, float* __restrict__ outf,
          int M, int N, int K, int ldc)
{
    int wid  = blockIdx.x * 8 + (threadIdx.x >> 5);
    int lane = threadIdx.x & 31;
    int nGroups = N >> 6;
    int tn4 = wid % nGroups;
    int tm  = wid / nGroups;
    if (tm >= (M >> 4)) return;
    int hf  = lane >> 4;
    int l16 = lane & 15;

    v8f acc[4];
#pragma unroll
    for (int t = 0; t < 4; t++) acc[t] = (v8f){};

    const u16* ap = A + (size_t)(tm * 16 + l16) * K + hf * 8;          // A frag: two 8-elem runs
    const u16* bp = Bt + (size_t)(tn4 * 64 + l16) * K + hf * 8;        // B frag: two 8-elem runs

    for (int k0 = 0; k0 < K; k0 += 32) {
        Frag fa;
        fa.q[0] = *(const uint4*)(ap + k0);        // K = k0 + hf*8 + 0..7
        fa.q[1] = *(const uint4*)(ap + k0 + 16);   // K = k0 + 16 + hf*8 + 0..7
#pragma unroll
        for (int t = 0; t < 4; t++) {
            const u16* b = bp + (size_t)t * 16 * K + k0;
            Frag fb;
            fb.q[0] = *(const uint4*)(b);          // K = k0 + hf*8 + 0..7
            fb.q[1] = *(const uint4*)(b + 16);     // K = k0 + 16 + hf*8 + 0..7
            acc[t] = __builtin_amdgcn_wmma_f32_16x16x32_bf16(
                false, fa.v, false, fb.v, (short)0, acc[t], false, false);
        }
    }

#pragma unroll
    for (int t = 0; t < 4; t++) {
        int col = tn4 * 64 + t * 16 + l16;
        float bv = bias ? bias[col] : 0.f;
#pragma unroll
        for (int r = 0; r < 8; r++) {
            int row = tm * 16 + hf * 8 + r;
            float v = acc[t][r] + bv;
            if (EPI == 0) {
                outb[(size_t)row * ldc + col] = f2bf(v);
            } else if (EPI == 1) {
                v = 0.5f * v * (1.f + erff(v * 0.70710678118654752f));
                outb[(size_t)row * ldc + col] = f2bf(v);
            } else if (EPI == 2) {
                int win = row / 49, n = row - win * 49;
                int b   = win >> 8, wi = win & 255;
                int hp  = (wi >> 4) * 7 + n / 7;
                int wp  = (wi & 15) * 7 + n % 7;
                int h = hp + 3; if (h >= 112) h -= 112;
                int w = wp + 3; if (w >= 112) w -= 112;
                size_t base = ((size_t)b * 12544 + h * 112 + w) * 192;
                outf[base + col] = resid[base + col] + v;
            } else { // EPI == 3, in-place residual
                size_t idx = (size_t)row * ldc + col;
                outf[idx] = resid[idx] + v;
            }
        }
    }
}

// ---------------------------------------------------------------------------
// Attention: 1 block (128 thr = 4 waves) per (window, head).
// S = Q K^T * 0.25 + rpb + shift-mask ; softmax ; O = P V. 49 padded to 64.
// ---------------------------------------------------------------------------
__global__ void __launch_bounds__(128)
attn_kernel(const u16* __restrict__ qkv, const float* __restrict__ rpb,
            u16* __restrict__ out)
{
    __shared__ __align__(16) float S[64][68];
    __shared__ __align__(16) u16   P[64][80];
    __shared__ __align__(16) u16   Vt[16][64];

    int bi   = blockIdx.x;
    int win  = bi / 12, head = bi - win * 12;
    int tid  = threadIdx.x;
    int wave = tid >> 5, lane = tid & 31;
    int hf   = lane >> 4, l16 = lane & 15;
    size_t wbase = (size_t)win * 49 * 576;

    // stage V^T (d, k) into LDS, zero-padded keys
    for (int idx = tid; idx < 16 * 64; idx += 128) {
        int d = idx >> 6, k = idx & 63;
        Vt[d][k] = (k < 49) ? qkv[wbase + (size_t)k * 576 + 384 + head * 16 + d]
                            : (u16)0;
    }

    // A fragment of Q for this wave's m-tile (K = HD = 16, padded to 32)
    Frag fq;
    int qrow = wave * 16 + l16;
    if (qrow < 49) {
        const u16* qp = qkv + wbase + (size_t)qrow * 576 + head * 16 + hf * 8;
        fq.q[0] = *(const uint4*)qp;
    } else {
        fq.q[0] = make_uint4(0, 0, 0, 0);
    }
    fq.q[1] = make_uint4(0, 0, 0, 0);   // K = 16..31 zero pad

    __syncthreads();

    // S tiles: wave handles m-tile = wave, all 4 n-tiles
    for (int nt = 0; nt < 4; nt++) {
        Frag fk;
        int krow = nt * 16 + l16;
        if (krow < 49) {     // B frag: elems 0..7 = K hf*8+0..7, elems 8..15 = K 16..31 -> zero pad
            const u16* kp = qkv + wbase + (size_t)krow * 576 + 192 + head * 16 + hf * 8;
            fk.q[0] = *(const uint4*)kp;
            fk.q[1] = make_uint4(0, 0, 0, 0);
        } else {
            fk.q[0] = make_uint4(0, 0, 0, 0);
            fk.q[1] = make_uint4(0, 0, 0, 0);
        }
        v8f s = (v8f){};
        s = __builtin_amdgcn_wmma_f32_16x16x32_bf16(
            false, fq.v, false, fk.v, (short)0, s, false, false);
#pragma unroll
        for (int r = 0; r < 8; r++)
            S[wave * 16 + hf * 8 + r][nt * 16 + l16] = s[r];
    }
    __syncthreads();

    // per-row softmax with rel-pos bias and shift mask (threads 0..63 own rows)
    if (tid < 64) {
        int i = tid;
        if (i < 49) {
            int wi = win & 255;
            int wh = wi >> 4, ww = wi & 15;
            int ir = i / 7, ic = i - ir * 7;
            int hi = wh * 7 + ir, wci = ww * 7 + ic;
            int regI = (hi < 105 ? 0 : (hi < 108 ? 1 : 2)) * 3 +
                       (wci < 105 ? 0 : (wci < 108 ? 1 : 2));
            float m = -1e30f;
            for (int j = 0; j < 49; j++) {
                int jr = j / 7, jc = j - jr * 7;
                int hj = wh * 7 + jr, wcj = ww * 7 + jc;
                int regJ = (hj < 105 ? 0 : (hj < 108 ? 1 : 2)) * 3 +
                           (wcj < 105 ? 0 : (wcj < 108 ? 1 : 2));
                int rel = (ir - jr + 6) * 13 + (ic - jc + 6);
                float v = S[i][j] * 0.25f + rpb[rel * 12 + head] +
                          (regI == regJ ? 0.f : -100.f);
                S[i][j] = v;
                m = fmaxf(m, v);
            }
            float sum = 0.f;
            for (int j = 0; j < 49; j++) {
                float e = __expf(S[i][j] - m);
                S[i][j] = e; sum += e;
            }
            float inv = 1.f / sum;
            for (int j = 0; j < 49; j++) P[i][j] = f2bf(S[i][j] * inv);
            for (int j = 49; j < 64; j++) P[i][j] = 0;
        } else {
            for (int j = 0; j < 64; j++) P[i][j] = 0;
        }
    }
    __syncthreads();

    // O = P(64x64 bf16) @ V(64x16): 2 K-steps of 32, one 16x16 tile per wave
    v8f o = (v8f){};
    int prow = wave * 16 + l16;
    const u16* pr = &P[prow][0];
    const u16* vr = &Vt[l16][0];
#pragma unroll
    for (int g = 0; g < 2; g++) {
        Frag fp, fv;
        fp.q[0] = *(const uint4*)(pr + g * 32 + hf * 8);
        fp.q[1] = *(const uint4*)(pr + g * 32 + 16 + hf * 8);
        fv.q[0] = *(const uint4*)(vr + g * 32 + hf * 8);
        fv.q[1] = *(const uint4*)(vr + g * 32 + 16 + hf * 8);
        o = __builtin_amdgcn_wmma_f32_16x16x32_bf16(
            false, fp.v, false, fv.v, (short)0, o, false, false);
    }
#pragma unroll
    for (int r = 0; r < 8; r++) {
        int token = wave * 16 + hf * 8 + r;
        if (token < 49)
            out[((size_t)win * 49 + token) * 192 + head * 16 + l16] = f2bf(o[r]);
    }
}

// ---------------------------------------------------------------------------
// Launcher
// ---------------------------------------------------------------------------
extern "C" void kernel_launch(void* const* d_in, const int* in_sizes, int n_in,
                              void* d_out, int out_size, void* d_ws, size_t ws_size,
                              hipStream_t stream)
{
    (void)in_sizes; (void)n_in; (void)out_size; (void)ws_size;

    const float* x   = (const float*)d_in[0];
    const float* n1g = (const float*)d_in[1];
    const float* n1b = (const float*)d_in[2];
    const float* qw  = (const float*)d_in[3];
    const float* qb  = (const float*)d_in[4];
    const float* kw  = (const float*)d_in[5];
    const float* kb  = (const float*)d_in[6];
    const float* vw  = (const float*)d_in[7];
    const float* vb  = (const float*)d_in[8];
    const float* pw  = (const float*)d_in[9];
    const float* pb  = (const float*)d_in[10];
    const float* rpb = (const float*)d_in[11];
    const float* n2g = (const float*)d_in[12];
    const float* n2b = (const float*)d_in[13];
    const float* f1w = (const float*)d_in[14];
    const float* f1b = (const float*)d_in[15];
    const float* f2w = (const float*)d_in[16];
    const float* f2b = (const float*)d_in[17];

    const int M = 200704;

    // workspace layout (bytes); requires ~387 MB
    char* ws = (char*)d_ws;
    const size_t XW_OFF  = 0;                          // 200704*192*2 = 77,070,336
    const size_t QKV_OFF = 77070336;                   // 200704*576*2 = 231,211,008
    const size_t ATT_OFF = QKV_OFF + 231211008;        // 77,070,336
    const size_t W_OFF   = ATT_OFF + 77070336;         // weights (bf16, transposed)

    u16* xw   = (u16*)(ws + XW_OFF);
    u16* qkv  = (u16*)(ws + QKV_OFF);
    u16* attn = (u16*)(ws + ATT_OFF);
    u16* wqT  = (u16*)(ws + W_OFF);                    // (192,192)
    u16* wkT  = wqT + 192 * 192;
    u16* wvT  = wkT + 192 * 192;
    u16* wpT  = wvT + 192 * 192;
    u16* w1T  = wpT + 192 * 192;                       // (768,192)
    u16* w2T  = w1T + 768 * 192;                       // (192,768)
    u16* hbuf = qkv;                                   // fc1 out (200704,768) reuses qkv+attn span
    u16* hin  = xw;                                    // ln2 out reuses xw
    float* x1 = (float*)d_out;                         // attn-branch result lives in d_out

    // 1) weights -> bf16 transposed (N,K)
    transpose_bf16<<<144, 256, 0, stream>>>(qw,  wqT, 192, 192);
    transpose_bf16<<<144, 256, 0, stream>>>(kw,  wkT, 192, 192);
    transpose_bf16<<<144, 256, 0, stream>>>(vw,  wvT, 192, 192);
    transpose_bf16<<<144, 256, 0, stream>>>(pw,  wpT, 192, 192);
    transpose_bf16<<<576, 256, 0, stream>>>(f1w, w1T, 192, 768);
    transpose_bf16<<<576, 256, 0, stream>>>(f2w, w2T, 768, 192);

    // 2) LN1 + shift + window partition
    ln1_shift_part<<<25088, 256, 0, stream>>>(x, n1g, n1b, xw);

    // 3) Q/K/V GEMMs into interleaved (row, 576) bf16
    gemm_wmma<0><<<4704, 256, 0, stream>>>(xw, wqT, qb, nullptr, qkv,       nullptr, M, 192, 192, 576);
    gemm_wmma<0><<<4704, 256, 0, stream>>>(xw, wkT, kb, nullptr, qkv + 192, nullptr, M, 192, 192, 576);
    gemm_wmma<0><<<4704, 256, 0, stream>>>(xw, wvT, vb, nullptr, qkv + 384, nullptr, M, 192, 192, 576);

    // 4) windowed attention (4096 windows x 12 heads)
    attn_kernel<<<49152, 128, 0, stream>>>(qkv, rpb, attn);

    // 5) proj GEMM + window-reverse/unshift scatter + residual -> x1 (= d_out)
    gemm_wmma<2><<<4704, 256, 0, stream>>>(attn, wpT, pb, x, nullptr, x1, M, 192, 192, 0);

    // 6) LN2 -> bf16
    ln2_kernel<<<25088, 256, 0, stream>>>(x1, n2g, n2b, hin);

    // 7) fc1 + GELU -> bf16
    gemm_wmma<1><<<18816, 256, 0, stream>>>(hin, w1T, f1b, nullptr, hbuf, nullptr, M, 768, 192, 768);

    // 8) fc2 + residual (in place on d_out)
    gemm_wmma<3><<<4704, 256, 0, stream>>>(hbuf, w2T, f2b, x1, nullptr, x1, M, 192, 768, 192);
}
